// Attention_70377334112631
// MI455X (gfx1250) — matrix operand order repaired and checked by reading, not compiled, not run on hardware
//
#include <hip/hip_runtime.h>
#include <hip/hip_bf16.h>
#include <cstdint>
#include <cstddef>

// Problem constants (match reference)
#define BB 4
#define NN 1024
#define MM 1024
#define HH 512
#define ITERS 100

typedef __attribute__((ext_vector_type(16))) _Float16 v16h;
typedef __attribute__((ext_vector_type(8)))  _Float16 v8h;
typedef __attribute__((ext_vector_type(4)))  _Float16 v4h;
typedef __attribute__((ext_vector_type(8)))  float    v8f;

// ---------------------------------------------------------------------------
// helpers
// ---------------------------------------------------------------------------
// online logsumexp combine: (m,s) <- (m,s) ⊕ (m2,s2)
__device__ __forceinline__ void lse_combine(float& m, float& s, float m2, float s2) {
    float mm = fmaxf(m, m2);
    s = s * __expf(m - mm) + s2 * __expf(m2 - mm);
    m = mm;
}

// ---------------------------------------------------------------------------
// fp32 -> fp16 bulk conversion (8 elements / thread, b128 in, b128 out)
// ---------------------------------------------------------------------------
__global__ __launch_bounds__(256)
void cvt_f16_kernel(const float* __restrict__ src, _Float16* __restrict__ dst) {
    const size_t q = (size_t)blockIdx.x * blockDim.x + threadIdx.x;
    const float4* s = (const float4*)src;
    float4 x0 = s[2 * q], x1 = s[2 * q + 1];
    v8h o;
    o[0] = (_Float16)x0.x; o[1] = (_Float16)x0.y;
    o[2] = (_Float16)x0.z; o[3] = (_Float16)x0.w;
    o[4] = (_Float16)x1.x; o[5] = (_Float16)x1.y;
    o[6] = (_Float16)x1.z; o[7] = (_Float16)x1.w;
    ((v8h*)dst)[q] = o;
}

// ---------------------------------------------------------------------------
// enc [B,M,H] f32 -> encTH [B,H,M] f16  (tiled LDS transpose + convert)
// ---------------------------------------------------------------------------
__global__ __launch_bounds__(256)
void transpose_enc_kernel(const float* __restrict__ enc, _Float16* __restrict__ encTH) {
    __shared__ float tile[32][33];
    const int b  = blockIdx.x;
    const int m0 = blockIdx.y * 32;
    const int h0 = blockIdx.z * 32;
    const int tx = threadIdx.x;     // 0..31
    const int ty = threadIdx.y;     // 0..7
    const float* src = enc + (size_t)b * MM * HH;
#pragma unroll
    for (int r = 0; r < 32; r += 8)
        tile[ty + r][tx] = src[(size_t)(m0 + ty + r) * HH + h0 + tx];
    __syncthreads();
    _Float16* dst = encTH + (size_t)b * HH * MM;
#pragma unroll
    for (int r = 0; r < 32; r += 8)
        dst[(size_t)(h0 + ty + r) * MM + m0 + tx] = (_Float16)tile[tx][ty + r];
}

// ---------------------------------------------------------------------------
// GEMM1: scores[b,n,m] = sum_h dec[b,n,h] * enc[b,m,h]   (f16 operands)
// one 16x16 output tile per wave32, K-loop of HH/32 WMMAs
// ---------------------------------------------------------------------------
__global__ __launch_bounds__(128)
void gemm_scores_kernel(const _Float16* __restrict__ decH,
                        const _Float16* __restrict__ encH,
                        float* __restrict__ scores) {
    const int wave = blockIdx.x * (blockDim.x >> 5) + (threadIdx.x >> 5);
    const int lane = threadIdx.x & 31;
    const int b  = wave >> 12;          // / (64*64)
    const int tn = (wave >> 6) & 63;    // N tile
    const int tm = wave & 63;           // M tile
    const int l16  = lane & 15;
    const int half = lane >> 4;

    // A (16x32 f16): lane half 0 holds K 0..15, half 1 holds K 16..31
    const _Float16* Arow = decH + ((size_t)b * NN + tn * 16 + l16) * HH + half * 16;
    // B (32x16 f16): lane half 0 holds K 0..15, half 1 holds K 16..31, col = l16
    const _Float16* Brow = encH + ((size_t)b * MM + tm * 16 + l16) * HH + half * 16;

    v8f acc = {};
    for (int kt = 0; kt < HH / 32; ++kt) {
        v8h alo = *(const v8h*)(Arow + kt * 32);
        v8h ahi = *(const v8h*)(Arow + kt * 32 + 8);
        v16h a  = __builtin_shufflevector(alo, ahi,
                   0, 1, 2, 3, 4, 5, 6, 7, 8, 9, 10, 11, 12, 13, 14, 15);
        v16h bv = *(const v16h*)(Brow + kt * 32);
        acc = __builtin_amdgcn_wmma_f32_16x16x32_f16(false, a, false, bv,
                                                     (short)0, acc, false, false);
    }
    // C/D layout: n = l16, m = half*8 + v
    float* dst = scores + ((size_t)b * NN + tn * 16 + half * 8) * MM + tm * 16 + l16;
#pragma unroll
    for (int v = 0; v < 8; ++v) dst[(size_t)v * MM] = acc[v];
}

// ---------------------------------------------------------------------------
// Sinkhorn potential updates.  la = scores + F_i + G_j  (never materialized)
// ---------------------------------------------------------------------------
__global__ __launch_bounds__(256)
void init_g_kernel(float* __restrict__ G) {
    G[blockIdx.x * 256 + threadIdx.x] = 0.0f;
}

// F_i = -LSE_j( scores[b,i,j] + G[b,j] )   — one block per row
__global__ __launch_bounds__(256)
void f_update_kernel(const float* __restrict__ scores,
                     const float* __restrict__ G,
                     float* __restrict__ F) {
    const int row = blockIdx.x;          // 0..B*N-1
    const int b   = row >> 10;           // / NN
    const float4* sr = (const float4*)(scores + (size_t)row * MM);
    const float4* g  = (const float4*)(G + (size_t)b * MM);
    const int t = threadIdx.x;

    float4 sv = sr[t];
    float4 gv = g[t];
    float m = sv.x + gv.x, s = 1.0f;
    lse_combine(m, s, sv.y + gv.y, 1.0f);
    lse_combine(m, s, sv.z + gv.z, 1.0f);
    lse_combine(m, s, sv.w + gv.w, 1.0f);

    __shared__ float sm[256], ss[256];
    sm[t] = m; ss[t] = s; __syncthreads();
#pragma unroll
    for (int d = 128; d >= 1; d >>= 1) {
        if (t < d) {
            float mm = sm[t], sv2 = ss[t];
            lse_combine(mm, sv2, sm[t + d], ss[t + d]);
            sm[t] = mm; ss[t] = sv2;
        }
        __syncthreads();
    }
    if (t == 0) F[row] = -(sm[0] + __logf(ss[0]));
}

// G_j = -LSE_i( scores[b,i,j] + F[b,i] )   — one block per 16 columns
__global__ __launch_bounds__(256)
void g_update_kernel(const float* __restrict__ scores,
                     const float* __restrict__ F,
                     float* __restrict__ G) {
    const int blk = blockIdx.x;          // 0..B*(M/16)-1
    const int b   = blk >> 6;            // / 64
    const int j0  = (blk & 63) << 4;
    const int t   = threadIdx.x;
    const int c   = t & 15;              // column within the 16
    const int ph  = t >> 4;              // row phase 0..15
    const float* sb = scores + (size_t)b * NN * MM + j0 + c;
    const float* f  = F + (size_t)b * NN;

    float m = -1e30f, s = 0.0f;
    for (int i = ph; i < NN; i += 16)
        lse_combine(m, s, sb[(size_t)i * MM] + f[i], 1.0f);

    __shared__ float sm[256], ss[256];
    sm[t] = m; ss[t] = s; __syncthreads();
#pragma unroll
    for (int d = 128; d >= 16; d >>= 1) {
        if (t < d) {
            float mm = sm[t], sv2 = ss[t];
            lse_combine(mm, sv2, sm[t + d], ss[t + d]);
            sm[t] = mm; ss[t] = sv2;
        }
        __syncthreads();
    }
    if (t < 16) G[(size_t)b * MM + j0 + t] = -(sm[t] + __logf(ss[t]));
}

// attn = exp(scores + F_i + G_j): writes f32 (output) and f16 (GEMM2 operand)
__global__ __launch_bounds__(256)
void attn_kernel(const float* __restrict__ scores,
                 const float* __restrict__ F,
                 const float* __restrict__ G,
                 float* __restrict__ attn,
                 _Float16* __restrict__ attnH) {
    const size_t q = (size_t)blockIdx.x * blockDim.x + threadIdx.x;
    const size_t e = q * 4;
    const int b = (int)(e >> 20);           // / (NN*MM)
    const int i = (int)((e >> 10) & 1023);
    const int j = (int)(e & 1023);
    float4 sv = *(const float4*)(scores + e);
    const float fi = F[(size_t)b * NN + i];
    const float* g = G + (size_t)b * MM + j;
    float4 o;
    o.x = __expf(sv.x + fi + g[0]);
    o.y = __expf(sv.y + fi + g[1]);
    o.z = __expf(sv.z + fi + g[2]);
    o.w = __expf(sv.w + fi + g[3]);
    *(float4*)(attn + e) = o;
    v4h oh;
    oh[0] = (_Float16)o.x; oh[1] = (_Float16)o.y;
    oh[2] = (_Float16)o.z; oh[3] = (_Float16)o.w;
    *(v4h*)(attnH + e) = oh;
}

// ---------------------------------------------------------------------------
// GEMM2: context[b,n,h] = sum_m attnH[b,n,m] * encTH[b,h,m]
// ---------------------------------------------------------------------------
__global__ __launch_bounds__(128)
void gemm_context_kernel(const _Float16* __restrict__ attnH,
                         const _Float16* __restrict__ encTH,
                         float* __restrict__ ctx) {
    const int wave = blockIdx.x * (blockDim.x >> 5) + (threadIdx.x >> 5);
    const int lane = threadIdx.x & 31;
    const int b  = wave >> 11;          // / (64*32)
    const int tn = (wave >> 5) & 63;    // N tile
    const int th = wave & 31;           // H tile
    const int l16  = lane & 15;
    const int half = lane >> 4;

    const _Float16* Arow = attnH + ((size_t)b * NN + tn * 16 + l16) * MM + half * 16;
    const _Float16* Brow = encTH + ((size_t)b * HH + th * 16 + l16) * MM + half * 16;

    v8f acc = {};
    for (int kt = 0; kt < MM / 32; ++kt) {
        v8h alo = *(const v8h*)(Arow + kt * 32);
        v8h ahi = *(const v8h*)(Arow + kt * 32 + 8);
        v16h a  = __builtin_shufflevector(alo, ahi,
                   0, 1, 2, 3, 4, 5, 6, 7, 8, 9, 10, 11, 12, 13, 14, 15);
        v16h bv = *(const v16h*)(Brow + kt * 32);
        acc = __builtin_amdgcn_wmma_f32_16x16x32_f16(false, a, false, bv,
                                                     (short)0, acc, false, false);
    }
    float* dst = ctx + ((size_t)b * NN + tn * 16 + half * 8) * HH + th * 16 + l16;
#pragma unroll
    for (int v = 0; v < 8; ++v) dst[(size_t)v * HH] = acc[v];
}

// ---------------------------------------------------------------------------
// launch
// ---------------------------------------------------------------------------
extern "C" void kernel_launch(void* const* d_in, const int* in_sizes, int n_in,
                              void* d_out, int out_size, void* d_ws, size_t ws_size,
                              hipStream_t stream) {
    (void)in_sizes; (void)n_in; (void)out_size; (void)ws_size;
    const float* enc = (const float*)d_in[0];   // [B,M,H]
    const float* dec = (const float*)d_in[1];   // [B,N,H]

    float* ctx  = (float*)d_out;                         // [B,N,H]
    float* attn = ctx + (size_t)BB * NN * HH;            // [B,N,M]

    // workspace carve (all sizes are multiples of 4 KiB -> alignment is safe)
    char* ws = (char*)d_ws;
    float*     scores = (float*)ws;                 ws += (size_t)BB * NN * MM * 4;  // 16 MiB
    float*     F      = (float*)ws;                 ws += (size_t)BB * NN * 4;
    float*     G      = (float*)ws;                 ws += (size_t)BB * MM * 4;
    _Float16*  decH   = (_Float16*)ws;              ws += (size_t)BB * NN * HH * 2;  // 4 MiB
    _Float16*  encH   = (_Float16*)ws;              ws += (size_t)BB * MM * HH * 2;  // 4 MiB
    _Float16*  encTH  = (_Float16*)ws;              ws += (size_t)BB * HH * MM * 2;  // 4 MiB
    _Float16*  attnH  = (_Float16*)ws;                                               // 8 MiB

    // one-time operand preparation (hoisted f32->f16 conversion)
    cvt_f16_kernel<<<(BB * NN * HH) / (8 * 256), 256, 0, stream>>>(dec, decH);
    cvt_f16_kernel<<<(BB * MM * HH) / (8 * 256), 256, 0, stream>>>(enc, encH);
    transpose_enc_kernel<<<dim3(BB, MM / 32, HH / 32), dim3(32, 8), 0, stream>>>(enc, encTH);

    gemm_scores_kernel<<<(BB * 64 * 64) / 4, 128, 0, stream>>>(decH, encH, scores);
    init_g_kernel<<<(BB * MM) / 256, 256, 0, stream>>>(G);

    for (int it = 0; it < ITERS; ++it) {
        f_update_kernel<<<BB * NN, 256, 0, stream>>>(scores, G, F);
        g_update_kernel<<<BB * (MM / 16), 256, 0, stream>>>(scores, F, G);
    }

    attn_kernel<<<(BB * NN * MM) / (4 * 256), 256, 0, stream>>>(scores, F, G, attn, attnH);
    gemm_context_kernel<<<(BB * 64 * 32) / 4, 128, 0, stream>>>(attnH, encTH, ctx);
}
